// MultiHeadAttention_37185826848780
// MI455X (gfx1250) — matrix operand order repaired and checked by reading, not compiled, not run on hardware
//
#include <hip/hip_runtime.h>
#include <hip/hip_bf16.h>

// ---------------- problem constants ----------------
#define BB     2
#define SS     2048
#define DD     768
#define HH     12
#define MM     40
#define DEPTH  64
#define SKV    2088      // S + M (real key count), divisible by 4
#define SKP    2112      // padded to 33 * 64
#define NKT    33        // SKP / 64
#define NROWS  (BB*SS)   // 4096

typedef __bf16 bf16_t;
typedef __attribute__((ext_vector_type(16))) __bf16 v16bf;
typedef __attribute__((ext_vector_type(8)))  __bf16 v8bf;
typedef __attribute__((ext_vector_type(8)))  float  v8f;
typedef __attribute__((ext_vector_type(4)))  unsigned int u32x4;
typedef __attribute__((ext_vector_type(8)))  unsigned int u32x8;

// ------------- WMMA fragment helpers (CDNA5 wave32 layouts) -------------
// A 16x32 bf16: lane%16 = M row; element e -> K = (e/8)*16 + (lane/16)*8 + e%8
__device__ inline v16bf frag_a(const bf16_t* rowp, int lh) {
  v8bf lo = *(const v8bf*)(rowp + lh*8);
  v8bf hi = *(const v8bf*)(rowp + 16 + lh*8);
  v16bf r;
#pragma unroll
  for (int i = 0; i < 8; ++i) { r[i] = lo[i]; r[i+8] = hi[i]; }
  return r;
}
// B 32x16 bf16: lane%16 = N col; element e -> K = (e/8)*16 + (lane/16)*8 + e%8
__device__ inline v16bf frag_b(const bf16_t* rowp, int lh) {
  v8bf lo = *(const v8bf*)(rowp + lh*8);
  v8bf hi = *(const v8bf*)(rowp + 16 + lh*8);
  v16bf r;
#pragma unroll
  for (int i = 0; i < 8; ++i) { r[i] = lo[i]; r[i+8] = hi[i]; }
  return r;
}
__device__ inline v8f wmma_bf16(v16bf a, v16bf b, v8f c) {
  return __builtin_amdgcn_wmma_f32_16x16x32_bf16(false, a, false, b,
                                                 (short)0, c, false, false);
}

// ------------- Tensor Data Mover: 2D bf16 tile -> LDS (with row padding) ----
// pad_interval codes: 3 = every 16 DWORDs (32 bf16), 4 = every 32 DWORDs (64 bf16)
// pad_amount fixed: code 3 = 4 DWORDs (8 bf16) -> LDS ld of 40 / 72 elements.
#define TDM_PI_16DW 3u
#define TDM_PI_32DW 4u
__device__ inline void tdm_load_2d_bf16(const bf16_t* lds_dst, const bf16_t* gsrc,
                                        unsigned tile_w, unsigned tile_h,
                                        unsigned row_stride_elems,
                                        unsigned pad_interval_code) {
  unsigned long long ga = (unsigned long long)(size_t)gsrc;
  unsigned lo = (unsigned)(size_t)lds_dst;           // LDS byte offset (flat low bits)
  u32x4 g0;
  g0[0] = 1u;                                        // count=1, user descriptor
  g0[1] = lo;                                        // lds_addr
  g0[2] = (unsigned)ga;                              // global_addr[31:0]
  g0[3] = (unsigned)((ga >> 32) & 0x01ffffffu) | (2u << 30);  // addr[56:32] | type=2
  u32x8 g1;
  g1[0] = (1u << 16)                                 // data_size = 2 bytes
        | (1u << 20)                                 // pad_enable
        | (pad_interval_code << 22)
        | (3u << 25);                                // pad_amount = 4 DWORDs
  g1[1] = (row_stride_elems & 0xffffu) << 16;        // tensor_dim0[15:0] = stride
  g1[2] = (row_stride_elems >> 16) & 0xffffu;        // tensor_dim0[31:16]; dim1 lo = 0
  g1[3] = 0x4000u | (tile_w << 16);                  // tensor_dim1 = 0x40000000; tile_dim0
  g1[4] = tile_h & 0xffffu;                          // tile_dim1; tile_dim2 = 0
  g1[5] = row_stride_elems;                          // tensor_dim0_stride[31:0]
  g1[6] = 0u;
  g1[7] = 0u;
  asm volatile("tensor_load_to_lds %0, %1" :: "s"(g0), "s"(g1) : "memory");
}
__device__ inline void tdm_wait() { __builtin_amdgcn_s_wait_tensorcnt(0); }

// ------------- K0: weight transpose + bf16 convert (once per call) -----------
__global__ __launch_bounds__(256)
void wconv_kernel(const float* __restrict__ W, bf16_t* __restrict__ WT) {
  int idx = blockIdx.x * 256 + threadIdx.x;
  if (idx < DD*DD) {
    int c = idx / DD, r = idx % DD;                  // WT[c][r], writes coalesced
    WT[idx] = (bf16_t)W[(size_t)r*DD + c];
  }
}

// ------------- K1: X(4096x768) @ Wq + bq -> bf16, scatter per mode -----------
// mode 0: qbuf  [bh][S][64]    mode 1: kbuf [bh][SKP][64]   mode 2: vbufT [bh][64][SKP]
__global__ __launch_bounds__(128)
void proj_gemm_kernel(const float* __restrict__ X, const bf16_t* __restrict__ WT,
                      const float* __restrict__ bias, bf16_t* __restrict__ out,
                      int mode) {
  __shared__ __align__(16) bf16_t As[64*40];
  __shared__ __align__(16) bf16_t Bs[64*40];        // [n][k], TDM-loaded from WT
  const int t = threadIdx.x, w = t >> 5, ln = t & 31, lh = ln >> 4, l16 = ln & 15;
  const int row0 = blockIdx.y * 64, col0 = blockIdx.x * 64;
  v8f acc[4] = {};
  for (int kk = 0; kk < DD; kk += 32) {
    if (t < 32)
      tdm_load_2d_bf16(Bs, &WT[(size_t)col0*DD + kk], 32, 64, DD, TDM_PI_16DW);
    {  // fp32 -> bf16 A tile, 2 threads/row, float4 loads + b128 LDS stores
      const int r = t >> 1, hp = t & 1;
      const float4* s4 = (const float4*)&X[(size_t)(row0 + r)*DD + kk + hp*16];
#pragma unroll
      for (int u = 0; u < 2; ++u) {
        float4 f0 = s4[2*u], f1 = s4[2*u + 1];
        v8bf bb;
        bb[0]=(bf16_t)f0.x; bb[1]=(bf16_t)f0.y; bb[2]=(bf16_t)f0.z; bb[3]=(bf16_t)f0.w;
        bb[4]=(bf16_t)f1.x; bb[5]=(bf16_t)f1.y; bb[6]=(bf16_t)f1.z; bb[7]=(bf16_t)f1.w;
        *(v8bf*)&As[r*40 + hp*16 + u*8] = bb;
      }
    }
    if (t < 32) tdm_wait();
    __syncthreads();
    v16bf a = frag_a(&As[(w*16 + l16)*40], lh);
#pragma unroll
    for (int n = 0; n < 4; ++n) {
      v16bf b = frag_b(&Bs[(n*16 + l16)*40], lh);
      acc[n] = wmma_bf16(a, b, acc[n]);
    }
    __syncthreads();
  }
#pragma unroll
  for (int n = 0; n < 4; ++n) {
    int cg = col0 + n*16 + l16;
    int h = cg >> 6, dep = cg & 63;
#pragma unroll
    for (int j = 0; j < 8; ++j) {
      int rg = row0 + w*16 + j + 8*lh;
      int b = rg >> 11, s = rg & 2047;
      int bh = b*HH + h;
      float vv = acc[n][j] + bias[cg];
      size_t idx;
      if (mode == 0)      idx = (((size_t)bh)*SS  + s)*64 + dep;
      else if (mode == 1) idx = (((size_t)bh)*SKP + s)*64 + dep;
      else                idx = (((size_t)bh)*64 + dep)*(size_t)SKP + s;
      out[idx] = (bf16_t)vv;
    }
  }
}

// ------------- K1b: persistent memory tokens -> kbuf rows / vbufT cols -------
__global__ void memtok_kernel(const float* __restrict__ mk, const float* __restrict__ mv,
                              bf16_t* __restrict__ kbuf, bf16_t* __restrict__ vbufT) {
  int bh = blockIdx.x, h = bh % HH;
  for (int i = threadIdx.x; i < 64*64; i += blockDim.x) {
    int kl = i >> 6, dep = i & 63;
    float kv = 0.f, vv = 0.f;
    if (kl < MM) {
      kv = 27.712812921102035f * mk[h*(MM*64) + kl*64 + dep];  // sqrt(768)
      vv = 6.324555320336759f  * mv[h*(MM*64) + kl*64 + dep];  // sqrt(40)
    }
    kbuf [(((size_t)bh)*SKP + 2048 + kl)*64 + dep] = (bf16_t)kv;
    vbufT[(((size_t)bh)*64 + dep)*(size_t)SKP + 2048 + kl] = (bf16_t)vv;
  }
}

// ------------- K2: exp(QK^T/8 + mask) tiles + partial row sums ---------------
__global__ __launch_bounds__(128)
void scores_kernel(const bf16_t* __restrict__ qbuf, const bf16_t* __restrict__ kbuf,
                   const float* __restrict__ mask, float* __restrict__ attn,
                   float* __restrict__ rowsumP) {
  __shared__ __align__(16) bf16_t qs[64*72];   // [qrow][depth], TDM-padded
  __shared__ __align__(16) bf16_t ks[64*72];   // [key][depth]  == BT[n][k]
  __shared__ float stile[64*65];
  const int t = threadIdx.x, w = t >> 5, ln = t & 31, lh = ln >> 4, l16 = ln & 15;
  const int kt = blockIdx.x, qt = blockIdx.y, bh = blockIdx.z;
  const int b = bh / HH, q0 = qt*64, k0 = kt*64;
  if (t < 32) {
    tdm_load_2d_bf16(qs, &qbuf[(((size_t)bh)*SS  + q0)*64], 64, 64, 64, TDM_PI_32DW);
    tdm_load_2d_bf16(ks, &kbuf[(((size_t)bh)*SKP + k0)*64], 64, 64, 64, TDM_PI_32DW);
    tdm_wait();
  }
  __syncthreads();
  v8f acc[4] = {};
#pragma unroll
  for (int kk = 0; kk < 64; kk += 32) {
    v16bf a = frag_a(&qs[(w*16 + l16)*72 + kk], lh);
#pragma unroll
    for (int n = 0; n < 4; ++n) {
      v16bf bfr = frag_b(&ks[(n*16 + l16)*72 + kk], lh);
      acc[n] = wmma_bf16(a, bfr, acc[n]);
    }
  }
  // scale + mask + exp -> LDS staging tile
#pragma unroll
  for (int n = 0; n < 4; ++n) {
    int col = n*16 + l16, gk = k0 + col;
    float mval = (gk < SS) ? mask[b*SS + gk] * (-1e9f) : 0.f;
    bool valid = gk < SKV;
#pragma unroll
    for (int j = 0; j < 8; ++j) {
      int r = w*16 + j + 8*lh;
      float vv = acc[n][j]*0.125f + mval;          // 1/sqrt(64)
      stile[r*65 + col] = valid ? __expf(vv) : 0.f;
    }
  }
  __syncthreads();
  // coalesced float4 stream of exp-scores to attn output region
  {
    const int r = t >> 1, hp = t & 1;
    size_t arow = (((size_t)bh)*SS + q0 + r)*SKV;
#pragma unroll
    for (int c4 = 0; c4 < 32; c4 += 4) {
      int c = hp*32 + c4, gk = k0 + c;
      if (gk < SKV) {
        float4 vv = { stile[r*65 + c],     stile[r*65 + c + 1],
                      stile[r*65 + c + 2], stile[r*65 + c + 3] };
        *(float4*)&attn[arow + gk] = vv;
      }
    }
  }
  // deterministic per-(row, ktile) sums (no float atomics)
  if (t < 64) {
    float s = 0.f;
#pragma unroll
    for (int c = 0; c < 64; ++c) s += stile[t*65 + c];
    rowsumP[(((size_t)bh)*SS + q0 + t)*NKT + kt] = s;
  }
}

// ------------- K3: normalize attn in place + ctx = attn @ V ------------------
__global__ __launch_bounds__(128)
void ctx_kernel(const bf16_t* __restrict__ vbufT, float* __restrict__ attn,
                const float* __restrict__ rowsumP, bf16_t* __restrict__ ctxbuf) {
  __shared__ __align__(16) bf16_t ps[64*72];   // probs [qrow][key]
  __shared__ __align__(16) bf16_t vT[64*72];   // [depth][key], TDM from vbufT
  __shared__ float rinv[64];
  const int t = threadIdx.x, w = t >> 5, ln = t & 31, lh = ln >> 4, l16 = ln & 15;
  const int qt = blockIdx.x, bh = blockIdx.y;
  const int b = bh / HH, h = bh % HH, q0 = qt*64;
  if (t < 64) {
    float s = 0.f;
    for (int k = 0; k < NKT; ++k)
      s += rowsumP[(((size_t)bh)*SS + q0 + t)*NKT + k];
    rinv[t] = 1.f / s;
  }
  __syncthreads();
  v8f acc[4] = {};
  for (int kt = 0; kt < NKT; ++kt) {
    int k0 = kt*64;
    if (t < 32)
      tdm_load_2d_bf16(vT, &vbufT[((size_t)bh*64)*SKP + k0], 64, 64, SKP, TDM_PI_32DW);
    {  // normalize attn (float4 in/out) + pack bf16 probs into LDS
      const int r = t >> 1, hp = t & 1;
      float rv = rinv[r];
      size_t arow = (((size_t)bh)*SS + q0 + r)*SKV;
#pragma unroll
      for (int c8 = 0; c8 < 32; c8 += 8) {
        int c = hp*32 + c8, gk = k0 + c;
        float4 pa = make_float4(0.f,0.f,0.f,0.f), pb = make_float4(0.f,0.f,0.f,0.f);
        if (gk < SKV) {
          float4* ap = (float4*)&attn[arow + gk];
          pa = *ap;
          pa.x *= rv; pa.y *= rv; pa.z *= rv; pa.w *= rv;
          *ap = pa;
        }
        if (gk + 4 < SKV) {
          float4* ap = (float4*)&attn[arow + gk + 4];
          pb = *ap;
          pb.x *= rv; pb.y *= rv; pb.z *= rv; pb.w *= rv;
          *ap = pb;
        }
        v8bf pk;
        pk[0]=(bf16_t)pa.x; pk[1]=(bf16_t)pa.y; pk[2]=(bf16_t)pa.z; pk[3]=(bf16_t)pa.w;
        pk[4]=(bf16_t)pb.x; pk[5]=(bf16_t)pb.y; pk[6]=(bf16_t)pb.z; pk[7]=(bf16_t)pb.w;
        *(v8bf*)&ps[r*72 + c] = pk;
      }
    }
    if (t < 32) tdm_wait();
    __syncthreads();
#pragma unroll
    for (int kk = 0; kk < 64; kk += 32) {
      v16bf a = frag_a(&ps[(w*16 + l16)*72 + kk], lh);
#pragma unroll
      for (int n = 0; n < 4; ++n) {
        v16bf bfr = frag_b(&vT[(n*16 + l16)*72 + kk], lh);
        acc[n] = wmma_bf16(a, bfr, acc[n]);
      }
    }
    __syncthreads();
  }
#pragma unroll
  for (int n = 0; n < 4; ++n) {
    int dep = n*16 + l16;
#pragma unroll
    for (int j = 0; j < 8; ++j) {
      int q = q0 + w*16 + j + 8*lh;
      ctxbuf[((size_t)(b*SS + q))*DD + h*64 + dep] = (bf16_t)acc[n][j];
    }
  }
}

// ------------- K4: ctx @ Wo + bo -> fp32 out ---------------------------------
__global__ __launch_bounds__(128)
void outproj_kernel(const bf16_t* __restrict__ ctxbuf, const bf16_t* __restrict__ WoT,
                    const float* __restrict__ bo, float* __restrict__ out) {
  __shared__ __align__(16) bf16_t As[64*40];
  __shared__ __align__(16) bf16_t Bs[64*40];
  const int t = threadIdx.x, w = t >> 5, ln = t & 31, lh = ln >> 4, l16 = ln & 15;
  const int row0 = blockIdx.y * 64, col0 = blockIdx.x * 64;
  v8f acc[4] = {};
  for (int kk = 0; kk < DD; kk += 32) {
    if (t < 32) {
      tdm_load_2d_bf16(As, &ctxbuf[(size_t)row0*DD + kk], 32, 64, DD, TDM_PI_16DW);
      tdm_load_2d_bf16(Bs, &WoT[(size_t)col0*DD + kk], 32, 64, DD, TDM_PI_16DW);
      tdm_wait();
    }
    __syncthreads();
    v16bf a = frag_a(&As[(w*16 + l16)*40], lh);
#pragma unroll
    for (int n = 0; n < 4; ++n) {
      v16bf bfr = frag_b(&Bs[(n*16 + l16)*40], lh);
      acc[n] = wmma_bf16(a, bfr, acc[n]);
    }
    __syncthreads();
  }
#pragma unroll
  for (int n = 0; n < 4; ++n) {
    int cg = col0 + n*16 + l16;
#pragma unroll
    for (int j = 0; j < 8; ++j) {
      int rg = row0 + w*16 + j + 8*lh;
      out[(size_t)rg*DD + cg] = acc[n][j] + bo[cg];
    }
  }
}

// ---------------- host-side orchestration ----------------
extern "C" void kernel_launch(void* const* d_in, const int* in_sizes, int n_in,
                              void* d_out, int out_size, void* d_ws, size_t ws_size,
                              hipStream_t stream) {
  (void)in_sizes; (void)n_in; (void)out_size; (void)ws_size;
  const float* v    = (const float*)d_in[0];
  const float* k    = (const float*)d_in[1];
  const float* q    = (const float*)d_in[2];
  const float* mask = (const float*)d_in[3];
  const float* Wq   = (const float*)d_in[4];
  const float* bq   = (const float*)d_in[5];
  const float* mk   = (const float*)d_in[6];
  const float* mv   = (const float*)d_in[7];
  const float* Wo   = (const float*)d_in[8];
  const float* bo   = (const float*)d_in[9];

  float* out  = (float*)d_out;                 // (B,S,D)
  float* attn = out + (size_t)BB*SS*DD;        // (B,H,S,SKV)

  char* ws = (char*)d_ws;
  size_t off = 0;
  bf16_t* qbuf   = (bf16_t*)(ws + off); off += (size_t)BB*HH*SS *64*sizeof(bf16_t);
  bf16_t* kbuf   = (bf16_t*)(ws + off); off += (size_t)BB*HH*SKP*64*sizeof(bf16_t);
  bf16_t* vbufT  = (bf16_t*)(ws + off); off += (size_t)BB*HH*64*SKP*sizeof(bf16_t);
  bf16_t* ctxbuf = (bf16_t*)(ws + off); off += (size_t)NROWS*DD*sizeof(bf16_t);
  bf16_t* WqT    = (bf16_t*)(ws + off); off += (size_t)DD*DD*sizeof(bf16_t);
  bf16_t* WoT    = (bf16_t*)(ws + off); off += (size_t)DD*DD*sizeof(bf16_t);
  float*  rowsumP = (float*)(ws + off); off += (size_t)BB*HH*SS*NKT*sizeof(float);

  dim3 blk(128);
  wconv_kernel<<<dim3((DD*DD + 255)/256), dim3(256), 0, stream>>>(Wq, WqT);
  wconv_kernel<<<dim3((DD*DD + 255)/256), dim3(256), 0, stream>>>(Wo, WoT);
  dim3 gproj(DD/64, NROWS/64);
  proj_gemm_kernel<<<gproj, blk, 0, stream>>>(q, WqT, bq, qbuf, 0);
  proj_gemm_kernel<<<gproj, blk, 0, stream>>>(k, WqT, bq, kbuf, 1);
  proj_gemm_kernel<<<gproj, blk, 0, stream>>>(v, WqT, bq, vbufT, 2);
  memtok_kernel<<<dim3(BB*HH), dim3(256), 0, stream>>>(mk, mv, kbuf, vbufT);
  scores_kernel<<<dim3(NKT, SS/64, BB*HH), blk, 0, stream>>>(qbuf, kbuf, mask, attn, rowsumP);
  ctx_kernel<<<dim3(SS/64, BB*HH), blk, 0, stream>>>(vbufT, attn, rowsumP, ctxbuf);
  outproj_kernel<<<dim3(DD/64, NROWS/64), blk, 0, stream>>>(ctxbuf, WoT, bo, out);
}
